// TreeLSTM_63153199121099
// MI455X (gfx1250) — matrix operand order repaired and checked by reading, not compiled, not run on hardware
//
#include <hip/hip_runtime.h>
#include <hip/hip_bf16.h>

// ---------------------------------------------------------------------------
// TreeLSTM on MI455X (gfx1250): bf16 WMMA GEMMs, fp32 accumulation.
// Internal levels: TDM (tensor_load_to_lds) DMAs the 32-row h_cat stripe into
// LDS (with hardware anti-bank-conflict padding), waves read A fragments from
// LDS, stream B (weights) from L2, and issue 2 WMMAs per B fragment.
// ---------------------------------------------------------------------------

typedef __attribute__((ext_vector_type(16))) __bf16 v16bf;
typedef __attribute__((ext_vector_type(8)))  __bf16 v8bf;
typedef __attribute__((ext_vector_type(8)))  float  v8f;
typedef __attribute__((ext_vector_type(4)))  unsigned int u32x4;
typedef __attribute__((ext_vector_type(8)))  int i32x8;
typedef __attribute__((ext_vector_type(4)))  int i32x4;

static constexpr int  B       = 256;
static constexpr int  NLEAF   = 256;
static constexpr int  H       = 512;
static constexpr int  X       = 300;
static constexpr int  XP      = 320;     // X padded to multiple of 32
static constexpr int  NIOU    = 1536;    // 3H
static constexpr int  NCAT    = 2560;    // 2H (f) + 3H (iou)
static constexpr int  KCAT    = 1024;    // 2H
static constexpr int  NODES   = 511;

// LDS A-stage geometry: TDM pads 16B every 1024B -> row stride 2080B.
static constexpr int  A_ROW_ELEMS = KCAT + 16;          // 1040 bf16 = 2080 B
static constexpr int  GSM_BYTES   = 16 * NCAT * 4;      // 160 KB epilogue tile
static constexpr int  ASTAGE_BYTES = 32 * 2080;         // 65 KB   (66560)
static constexpr int  SMEM_INTERNAL = GSM_BYTES + ASTAGE_BYTES;

__device__ __forceinline__ float sigf(float x) { return 1.0f / (1.0f + expf(-x)); }

__device__ __forceinline__ v16bf ld_afrag(const __bf16* p)
{
  v8bf lo = *(const v8bf*)(p);
  v8bf hi = *(const v8bf*)(p + 8);
  return __builtin_shufflevector(lo, hi,
      0, 1, 2, 3, 4, 5, 6, 7, 8, 9, 10, 11, 12, 13, 14, 15);
}

// ---------------------------------------------------------------------------
// Fragment layouts per CDNA5 ISA 7.12.2 (wave32):
//   A 16x32 bf16 : lane(0-15)=row, K {0..7,16..23}; lane(16-31)=row, K {8..15,24..31}
//   B 32x16 bf16 : lane&15 = column; here A and B both load 16 contiguous K per lane-half
//   C 16x16 f32  : lane&15 = column, VGPR j = row (lane>>4)*8 + j
// ---------------------------------------------------------------------------

// Leaf path: A direct from global. Two 16-row blocks share each B fragment.
template<int KDIM, int LDA, int LDW, int TPW>
__device__ __forceinline__ void wmma_2blocks_global(
    const __bf16* __restrict__ A, const __bf16* __restrict__ W,
    long rowBase, int colBase, v8f* acc0, v8f* acc1)
{
  const int lane = threadIdx.x & 31;
  const int m    = lane & 15;
  const int ksel = lane >> 4;
  const int n    = lane & 15;

  #pragma unroll
  for (int t = 0; t < TPW; ++t)
    #pragma unroll
    for (int e = 0; e < 8; ++e) { acc0[t][e] = 0.0f; acc1[t][e] = 0.0f; }

  const __bf16* __restrict__ arow0 = A + (rowBase + m) * (long)LDA + 16 * ksel;
  const __bf16* __restrict__ arow1 = arow0 + 16 * (long)LDA;

  for (int kk = 0; kk < KDIM; kk += 32) {
    const v16bf a0 = ld_afrag(arow0 + kk);
    const v16bf a1 = ld_afrag(arow1 + kk);
    #pragma unroll
    for (int t = 0; t < TPW; ++t) {
      const __bf16* wp = W + (long)(colBase + t * 16 + n) * LDW + kk + 16 * ksel;
      const v16bf b = *(const v16bf*)wp;
      acc0[t] = __builtin_amdgcn_wmma_f32_16x16x32_bf16(
          false, a0, false, b, (short)0, acc0[t], false, false);
      acc1[t] = __builtin_amdgcn_wmma_f32_16x16x32_bf16(
          false, a1, false, b, (short)0, acc1[t], false, false);
    }
  }
}

// Internal path: A fragments from the TDM-staged LDS tile (padded rows).
template<int TPW>
__device__ __forceinline__ void wmma_2blocks_lds(
    const __bf16* __restrict__ aStage,     // LDS, 32 rows x A_ROW_ELEMS
    const __bf16* __restrict__ W,
    int colBase, v8f* acc0, v8f* acc1)
{
  const int lane = threadIdx.x & 31;
  const int m    = lane & 15;
  const int ksel = lane >> 4;
  const int n    = lane & 15;

  #pragma unroll
  for (int t = 0; t < TPW; ++t)
    #pragma unroll
    for (int e = 0; e < 8; ++e) { acc0[t][e] = 0.0f; acc1[t][e] = 0.0f; }

  const __bf16* arow0 = aStage + m * A_ROW_ELEMS;
  const __bf16* arow1 = arow0 + 16 * A_ROW_ELEMS;

  for (int kk = 0; kk < KCAT; kk += 32) {
    const int k  = kk + 16 * ksel;
    const int kp = k + ((k >> 9) << 3);   // +8 elems (16B TDM pad) past 1024B
    const v16bf a0 = ld_afrag(arow0 + kp);
    const v16bf a1 = ld_afrag(arow1 + kp);
    #pragma unroll
    for (int t = 0; t < TPW; ++t) {
      const __bf16* wp = W + (long)(colBase + t * 16 + n) * KCAT + kk + 16 * ksel;
      const v16bf b = *(const v16bf*)wp;
      acc0[t] = __builtin_amdgcn_wmma_f32_16x16x32_bf16(
          false, a0, false, b, (short)0, acc0[t], false, false);
      acc1[t] = __builtin_amdgcn_wmma_f32_16x16x32_bf16(
          false, a1, false, b, (short)0, acc1[t], false, false);
    }
  }
}

// Spill one 16-row accumulator block to the LDS staging tile.
template<int TPW, int LDG>
__device__ __forceinline__ void store_block(const v8f* acc, int colBase,
                                            float* __restrict__ gsm)
{
  const int lane  = threadIdx.x & 31;
  const int n     = lane & 15;
  const int mBase = 8 * (lane >> 4);
  #pragma unroll
  for (int t = 0; t < TPW; ++t)
    #pragma unroll
    for (int j = 0; j < 8; ++j)
      gsm[(mBase + j) * LDG + colBase + t * 16 + n] = acc[t][j];
}

// TDM: DMA 64 KB from global `ga` into LDS offset
// `laddr`, 16B pad inserted every 1024B. D# per CDNA5 ISA 8.3/8.4.
__device__ __forceinline__ void tdm_load_a(unsigned long long ga, unsigned laddr)
{
  const unsigned units = 8192;                       // 64KB in 8-byte units
  u32x4 g0;
  g0[0] = 1u;                                        // count=1, user mode
  g0[1] = laddr;                                     // lds_addr (bytes)
  g0[2] = (unsigned)ga;                              // global_addr[31:0]
  g0[3] = (unsigned)((ga >> 32) & 0x01FFFFFFu)       // global_addr[56:32]
        | 0x80000000u;                               // type=2 ("image")
  i32x8 g1;
  g1[0] = (int)((3u << 16)                           // data_size = 8B
              | (1u << 20)                           // pad_enable
              | (7u << 22)                           // pad_interval: 256 DW
              | (3u << 25));                         // pad_amount:   4 DW
  g1[1] = (int)(units << 16);                        // tensor_dim0 lo16
  g1[2] = (int)(1u << 16);                           // tensor_dim1 = 1
  g1[3] = (int)(units << 16);                        // tile_dim0
  g1[4] = 1;                                         // tile_dim1 = 1
  g1[5] = (int)units;                                // tensor_dim0_stride
  g1[6] = 0;
  g1[7] = 0;
  i32x4 z4 = {0, 0, 0, 0};
#if defined(__clang_major__) && (__clang_major__ >= 23)
  i32x8 z8 = {0, 0, 0, 0, 0, 0, 0, 0};
  __builtin_amdgcn_tensor_load_to_lds(g0, g1, z4, z4, z8, 0);
#else
  __builtin_amdgcn_tensor_load_to_lds(g0, g1, z4, z4, 0);
#endif
  __builtin_amdgcn_s_wait_tensorcnt(0);
}

// --------------------------- weight prep kernels ---------------------------

__global__ void k_convert_leafW(const float* __restrict__ W, __bf16* __restrict__ out)
{
  const int nrow = blockIdx.x;           // 1536 rows
  const int k    = threadIdx.x;          // 320 threads
  out[(long)nrow * XP + k] = (k < X) ? (__bf16)W[(long)nrow * X + k] : (__bf16)0.0f;
}

__global__ void k_convert_catW(const float* __restrict__ Uf,
                               const float* __restrict__ Uiou,
                               __bf16* __restrict__ out)
{
  const int nrow = blockIdx.x;           // 2560 rows: [0,1024)=U_f, [1024,2560)=U_iou
  const float* src = (nrow < 1024) ? (Uf + (long)nrow * KCAT)
                                   : (Uiou + (long)(nrow - 1024) * KCAT);
  for (int k = threadIdx.x; k < KCAT; k += 256)
    out[(long)nrow * KCAT + k] = (__bf16)src[k];
}

__global__ void k_build_biascat(const float* __restrict__ Ufb,
                                const float* __restrict__ Uioub,
                                float* __restrict__ out)
{
  const int idx = blockIdx.x * 256 + threadIdx.x;  // 2560 total
  out[idx] = (idx < 1024) ? Ufb[idx] : Uioub[idx - 1024];
}

__global__ void k_gather_x(const int* __restrict__ wordid,
                           const float* __restrict__ emb,
                           __bf16* __restrict__ xb)
{
  const int r = blockIdx.x;              // 65536 leaf rows
  const int k = threadIdx.x;             // 320 threads
  const int wid = wordid[r];
  xb[(long)r * XP + k] = (k < X) ? (__bf16)emb[(long)wid * X + k] : (__bf16)0.0f;
}

// ------------------------------- leaf level --------------------------------
__device__ __forceinline__ void leaf_epilogue(
    const float* __restrict__ gsm, const float* __restrict__ bias,
    long rowBase, __bf16* __restrict__ hbOut, float* __restrict__ cOut)
{
  for (int idx = threadIdx.x; idx < 16 * H; idx += 512) {
    const int r = idx >> 9, nn = idx & (H - 1);
    const float* gr = gsm + r * NIOU;
    const float iv = gr[nn]            + bias[nn];
    const float ov = gr[H + nn]        + bias[H + nn];
    const float uv = gr[2 * H + nn]    + bias[2 * H + nn];
    const float c  = sigf(iv) * tanhf(uv);
    const float h  = sigf(ov) * tanhf(c);
    const long row = rowBase + r;
    cOut[row * H + nn]  = c;
    hbOut[row * H + nn] = (__bf16)h;
  }
}

__global__ __launch_bounds__(512)
void k_leaf_gemm(const __bf16* __restrict__ xb, const __bf16* __restrict__ WL,
                 const float* __restrict__ bias,
                 __bf16* __restrict__ hbOut, float* __restrict__ cOut)
{
  extern __shared__ float gsm[];                       // 16 x 1536 f32 = 96 KB
  const long rowBase = (long)blockIdx.x * 32;
  const int  wave    = threadIdx.x >> 5;               // 16 waves x 6 tiles
  const int  colBase = wave * 96;

  v8f acc0[6], acc1[6];
  wmma_2blocks_global<XP, XP, XP, 6>(xb, WL, rowBase, colBase, acc0, acc1);

  store_block<6, NIOU>(acc0, colBase, gsm);
  __syncthreads();
  leaf_epilogue(gsm, bias, rowBase, hbOut, cOut);
  __syncthreads();
  store_block<6, NIOU>(acc1, colBase, gsm);
  __syncthreads();
  leaf_epilogue(gsm, bias, rowBase + 16, hbOut, cOut);
}

// ----------------------------- internal level ------------------------------
__device__ __forceinline__ void internal_epilogue(
    const float* __restrict__ gsm, const float* __restrict__ bias,
    long rowBase, const float* __restrict__ cPrev,
    float* __restrict__ cOut, __bf16* __restrict__ hbOut)
{
  for (int idx = threadIdx.x; idx < 16 * H; idx += 512) {
    const int r = idx >> 9, nn = idx & (H - 1);
    const float* gr = gsm + r * NCAT;
    const float f0 = sigf(gr[nn]         + bias[nn]);
    const float f1 = sigf(gr[H + nn]     + bias[H + nn]);
    const float iv = gr[2 * H + nn]      + bias[2 * H + nn];
    const float ov = gr[3 * H + nn]      + bias[3 * H + nn];
    const float uv = gr[4 * H + nn]      + bias[4 * H + nn];
    const long row = rowBase + r;
    const float c0 = cPrev[(2 * row) * (long)H + nn];
    const float c1 = cPrev[(2 * row + 1) * (long)H + nn];
    const float c  = sigf(iv) * tanhf(uv) + f0 * c0 + f1 * c1;
    const float h  = sigf(ov) * tanhf(c);
    cOut[row * H + nn]  = c;
    hbOut[row * H + nn] = (__bf16)h;
  }
}

__global__ __launch_bounds__(512)
void k_internal_gemm(const __bf16* __restrict__ hbPrev,
                     const __bf16* __restrict__ Wcat,
                     const float* __restrict__ bias,
                     const float* __restrict__ cPrev,
                     float* __restrict__ cOut,
                     __bf16* __restrict__ hbOut)
{
  extern __shared__ char smem[];
  float*  gsm    = (float*)smem;                       // 16 x 2560 f32 = 160 KB
  __bf16* aStage = (__bf16*)(smem + GSM_BYTES);        // 32 x 2080 B

  const long rowBase = (long)blockIdx.x * 32;

  // Wave 0 DMAs the whole 32-row x 2KB h_cat stripe into LDS via TDM.
  if (threadIdx.x < 32) {
    const unsigned long long ga =
        (unsigned long long)hbPrev + (unsigned long long)rowBase * (KCAT * 2);
    tdm_load_a(ga, (unsigned)(unsigned long long)aStage);
  }
  __syncthreads();

  const int wave    = threadIdx.x >> 5;                // 16 waves x 10 tiles
  const int colBase = wave * 160;

  v8f acc0[10], acc1[10];
  wmma_2blocks_lds<10>(aStage, Wcat, colBase, acc0, acc1);

  store_block<10, NCAT>(acc0, colBase, gsm);
  __syncthreads();
  internal_epilogue(gsm, bias, rowBase, cPrev, cOut, hbOut);
  __syncthreads();
  store_block<10, NCAT>(acc1, colBase, gsm);
  __syncthreads();
  internal_epilogue(gsm, bias, rowBase + 16, cPrev, cOut, hbOut);
}

// ------------------------------- classifier --------------------------------
__global__ __launch_bounds__(256)
void k_logits(const __bf16* __restrict__ hb, const float* __restrict__ lw,
              const float* __restrict__ lb, float* __restrict__ out)
{
  const int idx = blockIdx.x * 256 + threadIdx.x;      // 256*511 exact
  const int b = idx / NODES, g = idx % NODES;
  const int offs[10] = {0, 256, 384, 448, 480, 496, 504, 508, 510, 511};
  int lvl = 0;
  while (g >= offs[lvl + 1]) ++lvl;
  const int  mlvl = NLEAF >> lvl;
  const long base = (long)B * offs[lvl] * H;
  const __bf16* hrow = hb + base + ((long)b * mlvl + (g - offs[lvl])) * H;
  float acc[5] = {lb[0], lb[1], lb[2], lb[3], lb[4]};
  for (int h = 0; h < H; ++h) {
    const float hv = (float)hrow[h];
    #pragma unroll
    for (int c = 0; c < 5; ++c) acc[c] += hv * lw[c * H + h];
  }
  float* op = out + ((long)b * NODES + g) * 5;
  #pragma unroll
  for (int c = 0; c < 5; ++c) op[c] = acc[c];
}

// ------------------------------ host launcher ------------------------------

static constexpr size_t OFF_WCAT = 0;                                    // 2560*1024 bf16
static constexpr size_t OFF_WL   = OFF_WCAT + (size_t)NCAT * KCAT * 2;   // 1536*320 bf16
static constexpr size_t OFF_BC   = OFF_WL   + (size_t)NIOU * XP * 2;     // 2560 f32
static constexpr size_t OFF_XB   = ((OFF_BC + (size_t)NCAT * 4 + 255) / 256) * 256;
static constexpr size_t OFF_HB   = ((OFF_XB + (size_t)B * NLEAF * XP * 2 + 255) / 256) * 256;
static constexpr size_t OFF_CA   = ((OFF_HB + (size_t)B * NODES * H * 2 + 255) / 256) * 256;
static constexpr size_t OFF_CB   = ((OFF_CA + (size_t)B * NLEAF * H * 4 + 255) / 256) * 256;

extern "C" void kernel_launch(void* const* d_in, const int* in_sizes, int n_in,
                              void* d_out, int out_size, void* d_ws, size_t ws_size,
                              hipStream_t stream)
{
  const int*   wordid  = (const int*)  d_in[0];
  const float* emb     = (const float*)d_in[1];
  const float* W_iou_w = (const float*)d_in[2];
  const float* W_iou_b = (const float*)d_in[3];
  const float* U_iou_w = (const float*)d_in[4];
  const float* U_iou_b = (const float*)d_in[5];
  const float* U_f_w   = (const float*)d_in[6];
  const float* U_f_b   = (const float*)d_in[7];
  const float* lin_w   = (const float*)d_in[8];
  const float* lin_b   = (const float*)d_in[9];
  float* out = (float*)d_out;

  char* ws = (char*)d_ws;
  __bf16* Wcat    = (__bf16*)(ws + OFF_WCAT);
  __bf16* WL      = (__bf16*)(ws + OFF_WL);
  float*  biasCat = (float*) (ws + OFF_BC);
  __bf16* xb      = (__bf16*)(ws + OFF_XB);
  __bf16* hb      = (__bf16*)(ws + OFF_HB);
  float*  cA      = (float*) (ws + OFF_CA);
  float*  cB      = (float*) (ws + OFF_CB);

  // Dynamic LDS above the 64 KB default (320 KB WGP pool on gfx1250).
  (void)hipFuncSetAttribute((const void*)k_leaf_gemm,
      hipFuncAttributeMaxDynamicSharedMemorySize, 16 * NIOU * 4);
  (void)hipFuncSetAttribute((const void*)k_internal_gemm,
      hipFuncAttributeMaxDynamicSharedMemorySize, SMEM_INTERNAL);

  // Precision prep (runs every call; deterministic).
  k_convert_leafW<<<NIOU, XP, 0, stream>>>(W_iou_w, WL);
  k_convert_catW<<<NCAT, 256, 0, stream>>>(U_f_w, U_iou_w, Wcat);
  k_build_biascat<<<10, 256, 0, stream>>>(U_f_b, U_iou_b, biasCat);
  k_gather_x<<<B * NLEAF, XP, 0, stream>>>(wordid, emb, xb);

  // Leaves: 65536 rows, 32 per workgroup.
  k_leaf_gemm<<<(B * NLEAF) / 32, 512, 16 * NIOU * 4, stream>>>(
      xb, WL, W_iou_b, hb, cA);

  // Internal levels (serial dependency chain, 8 halvings).
  const int offs[9] = {0, 256, 384, 448, 480, 496, 504, 508, 510};
  float* cbufs[2] = {cA, cB};
  for (int lvl = 1; lvl <= 8; ++lvl) {
    const int  m      = NLEAF >> lvl;
    const int  Mrows  = B * m;
    const long hbPrev = (long)B * offs[lvl - 1] * H;
    const long hbCur  = (long)B * offs[lvl] * H;
    k_internal_gemm<<<Mrows / 32, 512, SMEM_INTERNAL, stream>>>(
        hb + hbPrev, Wcat, biasCat,
        cbufs[(lvl + 1) & 1], cbufs[lvl & 1], hb + hbCur);
  }

  // Classifier over all 511*B nodes.
  k_logits<<<NODES, 256, 0, stream>>>(hb, lin_w, lin_b, out);

  (void)in_sizes; (void)n_in; (void)out_size; (void)ws_size;
}
